// DMM_6210522710230
// MI455X (gfx1250) — matrix operand order repaired and checked by reading, not compiled, not run on hardware
//
#include <hip/hip_runtime.h>
#include <hip/hip_bf16.h>

#define BB   256
#define TT   512
#define XD_  128
#define ZD_  64
#define ED_  256
#define TD_  256
#define RD_  512
#define G3_  (3 * RD_)   // 1536
#define GRUR 32          // batch rows per GRU workgroup

typedef __attribute__((ext_vector_type(16))) _Float16 v16h;
typedef __attribute__((ext_vector_type(8)))  _Float16 v8h;
typedef __attribute__((ext_vector_type(8)))  float    v8f;

__device__ __forceinline__ float sigm(float x) { return 1.f / (1.f + __expf(-x)); }

// ---- WMMA fragment loaders (CDNA5 16x16x32 f16 layouts, wave32) ----
// A (16xK tile, row-major, f16): lane m=lane&15; halves 0..7 = K kb..kb+7,
// halves 8..15 = K kb+16..kb+23, kb = (lane<16)?0:8.
__device__ __forceinline__ v16h frag_a(const _Float16* p0, int lda, int lane) {
  const int m  = lane & 15;
  const int kb = (lane & 16) ? 8 : 0;
  const _Float16* p = p0 + (size_t)m * lda + kb;
  v8h lo = *(const v8h*)(p);
  v8h hi = *(const v8h*)(p + 16);
  return __builtin_shufflevector(lo, hi, 0, 1, 2, 3, 4, 5, 6, 7,
                                 8, 9, 10, 11, 12, 13, 14, 15);
}
// B (Kx16 tile) from transposed (N-major) f16 weights: lane n=lane&15 holds
// K kb..kb+7 then kb+16..kb+23, kb = (lane<16)?0:8.
__device__ __forceinline__ v16h frag_b(const _Float16* bt0, int ldk, int lane) {
  const int n  = lane & 15;
  const int kb = (lane & 16) ? 8 : 0;
  const _Float16* p = bt0 + (size_t)n * ldk + kb;
  v8h lo = *(const v8h*)(p);
  v8h hi = *(const v8h*)(p + 16);
  return __builtin_shufflevector(lo, hi, 0, 1, 2, 3, 4, 5, 6, 7,
                                 8, 9, 10, 11, 12, 13, 14, 15);
}
#define WMMA(c, a, b)                                                        \
  (c) = __builtin_amdgcn_wmma_f32_16x16x32_f16(false, (a), false, (b),       \
                                               (short)0, (c), false, false)

// ---- weight convert + transpose: W f32 [K][N] -> WT f16 [N][K] ----
__global__ void k_convT(const float* __restrict__ W, _Float16* __restrict__ WT,
                        int K, int N) {
  int i = blockIdx.x * blockDim.x + threadIdx.x;
  if (i >= K * N) return;
  int k = i / N, n = i - k * N;
  WT[(size_t)n * K + k] = (_Float16)W[i];
}

// ---- GI = x_rev @ W_ih + b_ih : fully parallel WMMA GEMM over B*T rows ----
// grid = (B*T/16, 1536/128); block = 256 (8 waves, one 16x16 N-tile each).
__global__ __launch_bounds__(256) void k_gi(
    const float* __restrict__ x_rev,          // [B*T][128] (flattened)
    const _Float16* __restrict__ WihT,        // [1536][128]
    const float* __restrict__ b_ih,           // [1536]
    _Float16* __restrict__ gi)                // [B*T][1536]
{
  __shared__ _Float16 sXt[16 * XD_];
  const int tid = threadIdx.x, lane = tid & 31, wave = tid >> 5;
  const size_t r0 = (size_t)blockIdx.x * 16;   // 16 rows of B*T
  // stage 16x128 f32 -> f16 (8 floats/thread)
  {
    int e = tid * 8;
    int m = e >> 7, k = e & 127;
    const float4* p = (const float4*)(x_rev + (r0 + m) * XD_ + k);
    float4 v0 = p[0], v1 = p[1];
    _Float16* q = &sXt[m * XD_ + k];
    q[0] = (_Float16)v0.x; q[1] = (_Float16)v0.y;
    q[2] = (_Float16)v0.z; q[3] = (_Float16)v0.w;
    q[4] = (_Float16)v1.x; q[5] = (_Float16)v1.y;
    q[6] = (_Float16)v1.z; q[7] = (_Float16)v1.w;
  }
  __syncthreads();
  const int n0 = blockIdx.y * 128 + wave * 16;
  v8f acc = {};
#pragma unroll
  for (int kt = 0; kt < XD_; kt += 32)
    WMMA(acc, frag_a(sXt + kt, XD_, lane),
         frag_b(WihT + (size_t)n0 * XD_ + kt, XD_, lane));
  const int mb  = (lane & 16) ? 8 : 0;
  const int col = n0 + (lane & 15);
  const float bc = b_ih[col];
#pragma unroll
  for (int v = 0; v < 8; ++v)
    gi[(r0 + mb + v) * G3_ + col] = (_Float16)(acc[v] + bc);
}

// ---- GRU recurrence: 32 batch rows/WG, 16 waves, only h @ W_hh per step ----
// GI (with b_ih folded in) is consumed elementwise, so no loop-invariant
// B-fragments exist to hoist/spill. Each wave owns 2 hidden-column tiles; the
// 3 WhhT B-fragments per k-tile are reused across both 16-row M-tiles.
__global__ __launch_bounds__(512) void k_gru(
    const _Float16* __restrict__ gi,          // [B*T][1536]
    const float* __restrict__ b_hh,
    const _Float16* __restrict__ WhhT,        // [1536][512]
    _Float16* __restrict__ hout)              // [B][T][512]
{
  __shared__ _Float16 sH[GRUR * RD_];   // 32 KB
  const int tid = threadIdx.x, lane = tid & 31, wave = tid >> 5;
  const int b0 = blockIdx.x * GRUR;

  for (int i = tid; i < GRUR * RD_; i += 512) sH[i] = (_Float16)0.f;

  float bhr[2], bhz[2], bhn[2];
  const int nlo = lane & 15;
#pragma unroll
  for (int c = 0; c < 2; ++c) {
    int col = (wave * 2 + c) * 16 + nlo;
    bhr[c] = b_hh[col];
    bhz[c] = b_hh[RD_ + col];
    bhn[c] = b_hh[2 * RD_ + col];
  }
  __syncthreads();

  for (int t = 0; t < TT; ++t) {
    _Float16 hstash[2][16];          // [ct][mt*8+v] stays in VGPRs
#pragma unroll
    for (int c = 0; c < 2; ++c) {
      const int j0 = (wave * 2 + c) * 16;  // hidden column tile
      v8f ar0 = {}, ar1 = {}, az0 = {}, az1 = {}, ah0 = {}, ah1 = {};
      // hidden-hidden part: K = 512 (B fragments reused across both M-tiles)
#pragma unroll 4
      for (int kt = 0; kt < RD_; kt += 32) {
        v16h a0 = frag_a(sH + kt, RD_, lane);
        v16h a1 = frag_a(sH + 16 * RD_ + kt, RD_, lane);
        v16h br = frag_b(WhhT + (size_t)j0 * RD_ + kt, RD_, lane);
        v16h bz = frag_b(WhhT + (size_t)(RD_ + j0) * RD_ + kt, RD_, lane);
        v16h bn = frag_b(WhhT + (size_t)(2 * RD_ + j0) * RD_ + kt, RD_, lane);
        WMMA(ar0, a0, br); WMMA(ar1, a1, br);
        WMMA(az0, a0, bz); WMMA(az1, a1, bz);
        WMMA(ah0, a0, bn); WMMA(ah1, a1, bn);
      }
      const int mb  = (lane & 16) ? 8 : 0;
      const int col = j0 + (lane & 15);
#pragma unroll
      for (int mt = 0; mt < 2; ++mt) {
        const v8f& ar = mt ? ar1 : ar0;
        const v8f& az = mt ? az1 : az0;
        const v8f& ah = mt ? ah1 : ah0;
#pragma unroll
        for (int v = 0; v < 8; ++v) {
          int m = mt * 16 + mb + v;
          const _Float16* gp = gi + ((size_t)(b0 + m) * TT + t) * G3_;
          if (v == 0 && t + 1 < TT) __builtin_prefetch(gp + G3_, 0, 1);
          float gir = (float)gp[col];
          float giz = (float)gp[RD_ + col];
          float gin = (float)gp[2 * RD_ + col];
          float r  = sigm(ar[v] + gir + bhr[c]);
          float u  = sigm(az[v] + giz + bhz[c]);
          float nn = tanhf(gin + r * (ah[v] + bhn[c]));
          float ho = (float)sH[m * RD_ + col];     // read BEFORE barrier
          float hn = (1.f - u) * nn + u * ho;
          hstash[c][mt * 8 + v] = (_Float16)hn;
          // flip time back: h_rnn[:, T-1-t]
          hout[((size_t)(b0 + m) * TT + (TT - 1 - t)) * RD_ + col] =
              (_Float16)hn;
        }
      }
    }
    __syncthreads();   // all reads of sH complete
#pragma unroll
    for (int c = 0; c < 2; ++c) {
      const int j0  = (wave * 2 + c) * 16;
      const int mb  = (lane & 16) ? 8 : 0;
      const int col = j0 + (lane & 15);
#pragma unroll
      for (int mt = 0; mt < 2; ++mt)
#pragma unroll
        for (int v = 0; v < 8; ++v)
          sH[(mt * 16 + mb + v) * RD_ + col] = hstash[c][mt * 8 + v];
    }
    __syncthreads();
  }
}

// ---- DMM scan: 16 batch rows per workgroup, full T loop in-kernel ----
__global__ __launch_bounds__(256) void k_dmm(
    const float* __restrict__ eps_q, const float* __restrict__ eps_p,
    const _Float16* __restrict__ hws,
    const float* __restrict__ bzh, const float* __restrict__ bmu,
    const float* __restrict__ blv, const float* __restrict__ bg1,
    const float* __restrict__ bg2, const float* __restrict__ bp1,
    const float* __restrict__ bp2, const float* __restrict__ bmt,
    const float* __restrict__ blt, const float* __restrict__ be1,
    const float* __restrict__ be2, const float* __restrict__ be3,
    const float* __restrict__ mu_p0, const float* __restrict__ lv_p0,
    const _Float16* __restrict__ WzhT, const _Float16* __restrict__ WmuT,
    const _Float16* __restrict__ WlvT, const _Float16* __restrict__ Wg1T,
    const _Float16* __restrict__ Wg2T, const _Float16* __restrict__ Wp1T,
    const _Float16* __restrict__ Wp2T, const _Float16* __restrict__ WmtT,
    const _Float16* __restrict__ WltT, const _Float16* __restrict__ We1T,
    const _Float16* __restrict__ We2T, const _Float16* __restrict__ We3T,
    float* __restrict__ out_x, float* __restrict__ out_zt,
    float* __restrict__ out_zp, float* __restrict__ out_muq,
    float* __restrict__ out_lvq, float* __restrict__ out_mup,
    float* __restrict__ out_lvp)
{
  __shared__ _Float16 sZ[16 * ZD_];
  __shared__ _Float16 sA[16 * TD_];
  __shared__ _Float16 sB[16 * TD_];
  __shared__ _Float16 sHC[16 * RD_];
  __shared__ _Float16 sPr16[16 * ZD_];
  __shared__ float sG[16 * ZD_], sProp[16 * ZD_], sMuP[16 * ZD_],
      sLvP[16 * ZD_], sMuQ[16 * ZD_], sLvQ[16 * ZD_];

  const int tid = threadIdx.x, lane = tid & 31, wave = tid >> 5;
  const int b0 = blockIdx.x * 16;
  const int mb = (lane & 16) ? 8 : 0;

  for (int i = tid; i < 16 * ZD_; i += 256) sZ[i] = (_Float16)0.f;  // z0 = 0
  __syncthreads();

  for (int t = 0; t < TT; ++t) {
    // ---- phase 1: g1 = relu(z@Wg1+bg1)  /  p1 = relu(z@Wp1+bp1) ----
    {
      const bool gp = wave < 4;
      const _Float16* WT = gp ? Wg1T : Wp1T;
      const float* bias  = gp ? bg1 : bp1;
      _Float16* dst      = gp ? sA : sB;
      const int w4 = wave & 3;
#pragma unroll
      for (int i = 0; i < 4; ++i) {
        int n0 = (w4 * 4 + i) * 16;
        v8f acc = {};
#pragma unroll
        for (int kt = 0; kt < ZD_; kt += 32)
          WMMA(acc, frag_a(sZ + kt, ZD_, lane),
               frag_b(WT + (size_t)n0 * ZD_ + kt, ZD_, lane));
        int col = n0 + (lane & 15);
        float bc = bias[col];
#pragma unroll
        for (int v = 0; v < 8; ++v)
          dst[(mb + v) * TD_ + col] = (_Float16)fmaxf(acc[v] + bc, 0.f);
      }
    }
    __syncthreads();
    // ---- phase 2: g = sigmoid(g1@Wg2+bg2)  /  prop = p1@Wp2+bp2 ----
    {
      const bool gp = wave < 4;
      const _Float16* AA = gp ? sA : sB;
      const _Float16* WT = gp ? Wg2T : Wp2T;
      const float* bias  = gp ? bg2 : bp2;
      int n0 = (wave & 3) * 16;
      v8f acc = {};
#pragma unroll
      for (int kt = 0; kt < TD_; kt += 32)
        WMMA(acc, frag_a(AA + kt, TD_, lane),
             frag_b(WT + (size_t)n0 * TD_ + kt, TD_, lane));
      int col = n0 + (lane & 15);
      float bc = bias[col];
#pragma unroll
      for (int v = 0; v < 8; ++v) {
        float x = acc[v] + bc;
        int m = mb + v;
        if (gp) sG[m * ZD_ + col] = sigm(x);
        else { sProp[m * ZD_ + col] = x; sPr16[m * ZD_ + col] = (_Float16)fmaxf(x, 0.f); }
      }
    }
    __syncthreads();
    // ---- phase 3: mu_tr / lv_tr (with t==0 learned prior override) ----
    {
      const bool mp = wave < 4;
      const _Float16* AA = mp ? sZ : sPr16;
      const _Float16* WT = mp ? WmtT : WltT;
      int n0 = (wave & 3) * 16;
      v8f acc = {};
#pragma unroll
      for (int kt = 0; kt < ZD_; kt += 32)
        WMMA(acc, frag_a(AA + kt, ZD_, lane),
             frag_b(WT + (size_t)n0 * ZD_ + kt, ZD_, lane));
      int col = n0 + (lane & 15);
      if (mp) {
        float bc = bmt[col], p0 = mu_p0[col];
#pragma unroll
        for (int v = 0; v < 8; ++v) {
          int m = mb + v;
          float g = sG[m * ZD_ + col], pr = sProp[m * ZD_ + col];
          float mu = (1.f - g) * (acc[v] + bc) + g * pr;
          sMuP[m * ZD_ + col] = (t == 0) ? p0 : mu;
        }
      } else {
        float bc = blt[col], p0 = lv_p0[col];
#pragma unroll
        for (int v = 0; v < 8; ++v)
          sLvP[(mb + v) * ZD_ + col] = (t == 0) ? p0 : (acc[v] + bc);
      }
    }
    __syncthreads();
    // ---- phase 4: hc = 0.5*(tanh(z@Wzh+bzh) + h_t) ----
#pragma unroll
    for (int i = 0; i < 4; ++i) {
      int n0 = (wave * 4 + i) * 16;
      v8f acc = {};
#pragma unroll
      for (int kt = 0; kt < ZD_; kt += 32)
        WMMA(acc, frag_a(sZ + kt, ZD_, lane),
             frag_b(WzhT + (size_t)n0 * ZD_ + kt, ZD_, lane));
      int col = n0 + (lane & 15);
      float bc = bzh[col];
#pragma unroll
      for (int v = 0; v < 8; ++v) {
        int m = mb + v;
        float zh = tanhf(acc[v] + bc);
        float h = (float)hws[((size_t)(b0 + m) * TT + t) * RD_ + col];
        sHC[m * RD_ + col] = (_Float16)(0.5f * (zh + h));
      }
    }
    __syncthreads();
    // ---- phase 5: mu_q / lv_q = hc@Wmu|Wlv + b ----
    {
      const bool mq = wave < 4;
      const _Float16* WT = mq ? WmuT : WlvT;
      const float* bias  = mq ? bmu : blv;
      float* dst = mq ? sMuQ : sLvQ;
      int n0 = (wave & 3) * 16;
      v8f acc = {};
#pragma unroll 4
      for (int kt = 0; kt < RD_; kt += 32)
        WMMA(acc, frag_a(sHC + kt, RD_, lane),
             frag_b(WT + (size_t)n0 * RD_ + kt, RD_, lane));
      int col = n0 + (lane & 15);
      float bc = bias[col];
#pragma unroll
      for (int v = 0; v < 8; ++v) dst[(mb + v) * ZD_ + col] = acc[v] + bc;
    }
    __syncthreads();
    // ---- phase 6: reparameterize + write latent outputs ----
#pragma unroll
    for (int q = 0; q < 4; ++q) {
      int e = tid * 4 + q;            // 0..1023 over [16][64]
      int m = e >> 6, c = e & 63;
      size_t g = ((size_t)(b0 + m) * TT + t) * ZD_ + c;
      float muq = sMuQ[e], lvq = sLvQ[e], mup = sMuP[e], lvp = sLvP[e];
      float zt = muq + eps_q[g] * __expf(0.5f * lvq);
      float zp = mup + eps_p[g] * __expf(0.5f * lvp);
      out_zt[g] = zt; out_zp[g] = zp;
      out_muq[g] = muq; out_lvq[g] = lvq;
      out_mup[g] = mup; out_lvp[g] = lvp;
      sZ[e] = (_Float16)zt;           // z_prev for next step
    }
    __syncthreads();
    // ---- phase 7: e1 = relu(z_t@We1+be1) -> sA ----
#pragma unroll
    for (int i = 0; i < 2; ++i) {
      int n0 = (wave * 2 + i) * 16;
      v8f acc = {};
#pragma unroll
      for (int kt = 0; kt < ZD_; kt += 32)
        WMMA(acc, frag_a(sZ + kt, ZD_, lane),
             frag_b(We1T + (size_t)n0 * ZD_ + kt, ZD_, lane));
      int col = n0 + (lane & 15);
      float bc = be1[col];
#pragma unroll
      for (int v = 0; v < 8; ++v)
        sA[(mb + v) * TD_ + col] = (_Float16)fmaxf(acc[v] + bc, 0.f);
    }
    __syncthreads();
    // ---- phase 8: e2 = relu(e1@We2+be2) -> sB ----
#pragma unroll
    for (int i = 0; i < 2; ++i) {
      int n0 = (wave * 2 + i) * 16;
      v8f acc = {};
#pragma unroll
      for (int kt = 0; kt < ED_; kt += 32)
        WMMA(acc, frag_a(sA + kt, TD_, lane),
             frag_b(We2T + (size_t)n0 * ED_ + kt, ED_, lane));
      int col = n0 + (lane & 15);
      float bc = be2[col];
#pragma unroll
      for (int v = 0; v < 8; ++v)
        sB[(mb + v) * TD_ + col] = (_Float16)fmaxf(acc[v] + bc, 0.f);
    }
    __syncthreads();
    // ---- phase 9: x_t = sigmoid(e2@We3+be3) -> global ----
    {
      int n0 = wave * 16;
      v8f acc = {};
#pragma unroll
      for (int kt = 0; kt < ED_; kt += 32)
        WMMA(acc, frag_a(sB + kt, TD_, lane),
             frag_b(We3T + (size_t)n0 * ED_ + kt, ED_, lane));
      int col = n0 + (lane & 15);
      float bc = be3[col];
#pragma unroll
      for (int v = 0; v < 8; ++v) {
        int m = mb + v;
        out_x[((size_t)(b0 + m) * TT + t) * XD_ + col] = sigm(acc[v] + bc);
      }
    }
    __syncthreads();
  }
}

extern "C" void kernel_launch(void* const* d_in, const int* in_sizes, int n_in,
                              void* d_out, int out_size, void* d_ws,
                              size_t ws_size, hipStream_t stream) {
  (void)in_sizes; (void)n_in; (void)out_size; (void)ws_size;
  const float* x_rev = (const float*)d_in[1];
  const float* eps_q = (const float*)d_in[3];
  const float* eps_p = (const float*)d_in[4];
  const float* W_ih = (const float*)d_in[5];
  const float* W_hh = (const float*)d_in[6];
  const float* b_ih = (const float*)d_in[7];
  const float* b_hh = (const float*)d_in[8];
  const float* Wzh = (const float*)d_in[9],  *bzh = (const float*)d_in[10];
  const float* Wmu = (const float*)d_in[11], *bmu = (const float*)d_in[12];
  const float* Wlv = (const float*)d_in[13], *blv = (const float*)d_in[14];
  const float* Wg1 = (const float*)d_in[15], *bg1 = (const float*)d_in[16];
  const float* Wg2 = (const float*)d_in[17], *bg2 = (const float*)d_in[18];
  const float* Wp1 = (const float*)d_in[19], *bp1 = (const float*)d_in[20];
  const float* Wp2 = (const float*)d_in[21], *bp2 = (const float*)d_in[22];
  const float* Wmt = (const float*)d_in[23], *bmt = (const float*)d_in[24];
  const float* Wlt = (const float*)d_in[25], *blt = (const float*)d_in[26];
  const float* We1 = (const float*)d_in[27], *be1 = (const float*)d_in[28];
  const float* We2 = (const float*)d_in[29], *be2 = (const float*)d_in[30];
  const float* We3 = (const float*)d_in[31], *be3 = (const float*)d_in[32];
  const float* mu_p0 = (const float*)d_in[33], *lv_p0 = (const float*)d_in[34];

  _Float16* w = (_Float16*)d_ws;
  _Float16* WihT = w; w += (size_t)G3_ * XD_;
  _Float16* WhhT = w; w += (size_t)G3_ * RD_;
  _Float16* WzhT = w; w += (size_t)RD_ * ZD_;
  _Float16* WmuT = w; w += (size_t)ZD_ * RD_;
  _Float16* WlvT = w; w += (size_t)ZD_ * RD_;
  _Float16* Wg1T = w; w += (size_t)TD_ * ZD_;
  _Float16* Wg2T = w; w += (size_t)ZD_ * TD_;
  _Float16* Wp1T = w; w += (size_t)TD_ * ZD_;
  _Float16* Wp2T = w; w += (size_t)ZD_ * TD_;
  _Float16* WmtT = w; w += (size_t)ZD_ * ZD_;
  _Float16* WltT = w; w += (size_t)ZD_ * ZD_;
  _Float16* We1T = w; w += (size_t)ED_ * ZD_;
  _Float16* We2T = w; w += (size_t)ED_ * ED_;
  _Float16* We3T = w; w += (size_t)XD_ * ED_;
  _Float16* hws  = w; w += (size_t)BB * TT * RD_;  // [B][T][512] f16
  _Float16* giws = w;                              // [B*T][1536] f16

  auto conv = [&](const float* W, _Float16* WT, int K, int N) {
    int total = K * N;
    k_convT<<<(total + 255) / 256, 256, 0, stream>>>(W, WT, K, N);
  };
  conv(W_ih, WihT, XD_, G3_);
  conv(W_hh, WhhT, RD_, G3_);
  conv(Wzh, WzhT, ZD_, RD_);
  conv(Wmu, WmuT, RD_, ZD_);
  conv(Wlv, WlvT, RD_, ZD_);
  conv(Wg1, Wg1T, ZD_, TD_);
  conv(Wg2, Wg2T, TD_, ZD_);
  conv(Wp1, Wp1T, ZD_, TD_);
  conv(Wp2, Wp2T, TD_, ZD_);
  conv(Wmt, WmtT, ZD_, ZD_);
  conv(Wlt, WltT, ZD_, ZD_);
  conv(We1, We1T, ZD_, ED_);
  conv(We2, We2T, ED_, ED_);
  conv(We3, We3T, ED_, XD_);

  // fully parallel input GEMM over all B*T rows
  dim3 gi_grid((BB * TT) / 16, G3_ / 128);
  k_gi<<<gi_grid, 256, 0, stream>>>(x_rev, WihT, b_ih, giws);

  // sequential recurrence (hidden-hidden only)
  k_gru<<<BB / GRUR, 512, 0, stream>>>(giws, b_hh, WhhT, hws);

  float* o = (float*)d_out;
  float* out_x  = o;
  float* out_zt = o + (size_t)BB * TT * XD_;
  float* out_zp  = out_zt + (size_t)BB * TT * ZD_;
  float* out_muq = out_zp + (size_t)BB * TT * ZD_;
  float* out_lvq = out_muq + (size_t)BB * TT * ZD_;
  float* out_mup = out_lvq + (size_t)BB * TT * ZD_;
  float* out_lvp = out_mup + (size_t)BB * TT * ZD_;

  k_dmm<<<BB / 16, 256, 0, stream>>>(
      eps_q, eps_p, hws, bzh, bmu, blv, bg1, bg2, bp1, bp2, bmt, blt, be1,
      be2, be3, mu_p0, lv_p0, WzhT, WmuT, WlvT, Wg1T, Wg2T, Wp1T, Wp2T, WmtT,
      WltT, We1T, We2T, We3T, out_x, out_zt, out_zp, out_muq, out_lvq,
      out_mup, out_lvp);
}
